// Mamba_25022479466699
// MI455X (gfx1250) — compile-verified, matrix operand order checked by reading, not run on hardware
//
#include <hip/hip_runtime.h>
#include <hip/hip_bf16.h>
#include <math.h>

// ---------------- model dims ----------------
#define LAYERS   4
#define DMODEL   256
#define DINNER   512
#define DSTATE   16
#define NHEADS   8
#define HEADDIM  64
#define DCONV    4
#define DINT     256
#define NCLS     10
#define INFEAT   64
#define TLEN     1024
#define BATCH    8
#define DINPROJ  1064                 // 2*512 + 2*16 + 8
#define CONVDIM  544                  // 512 + 32
#define NROW     (BATCH * TLEN)       // 8192
#define EPSV     1e-5f

typedef __attribute__((ext_vector_type(16))) _Float16 v16h;
typedef __attribute__((ext_vector_type(8)))  _Float16 v8h;
typedef __attribute__((ext_vector_type(8)))  float    v8f;
typedef __attribute__((ext_vector_type(4)))  float    f4;

__device__ __forceinline__ float siluf(float x) { return x / (1.0f + __expf(-x)); }

// ========================================================================
// fp32 -> f16 bulk convert (vectorized, n multiple of 4)
// ========================================================================
__global__ __launch_bounds__(256)
void k_cvt16(const float* __restrict__ src, _Float16* __restrict__ dst, int n)
{
    int i4 = (blockIdx.x * blockDim.x + threadIdx.x) * 4;
    if (i4 >= n) return;
    f4 v = *(const f4*)(src + i4);
    dst[i4 + 0] = (_Float16)v.x;
    dst[i4 + 1] = (_Float16)v.y;
    dst[i4 + 2] = (_Float16)v.z;
    dst[i4 + 3] = (_Float16)v.w;
}

// ========================================================================
// WMMA GEMM:  C[M,N] = A[M,K] (f16 row-major) * W[N,K]^T (f16) + bias(f32)
// 64x64 C tile / 256 threads (8 waves); double-buffered LDS; register
// prefetch of next K tile; v_wmma_f32_16x16x32_f16 (fp32 accumulate).
// OOB B rows are CLAMPED (not predicated): a B row only feeds its own C
// column, and OOB columns are masked at the C store, so no zero-fill or
// exec-mask juggling is needed in the inner loop.
// ========================================================================
__global__ __launch_bounds__(256)
void k_gemm_wmma(const _Float16* __restrict__ A, const _Float16* __restrict__ W,
                 const float* __restrict__ bias, float* __restrict__ C,
                 int M, int N, int K)
{
    __shared__ _Float16 As[2][64 * 32];
    __shared__ _Float16 Bs[2][64 * 32];

    const int tid  = threadIdx.x;
    const int lane = tid & 31;
    const int wave = tid >> 5;
    const int m0 = blockIdx.y * 64;
    const int n0 = blockIdx.x * 64;
    const int tr  = wave & 3;   // tile row   (0..3)
    const int tcg = wave >> 2;  // col group  (0..1)

    // staging: thread -> (row = tid>>2, kcol = (tid&3)*8); one v8h per tile
    const int    srow = tid >> 2;           // 0..63
    const int    scol = (tid & 3) * 8;      // 0,8,16,24
    const int    brow = (n0 + srow < N) ? (n0 + srow) : (N - 1);   // clamp
    const size_t aoff = (size_t)(m0 + srow) * K + scol;
    const size_t boff = (size_t)brow * K + scol;
    const int    sidx = tid * 8;

    v8h ra = *(const v8h*)(A + aoff);
    v8h rb = *(const v8h*)(W + boff);
    *(v8h*)(&As[0][sidx]) = ra;
    *(v8h*)(&Bs[0][sidx]) = rb;

    v8f acc0 = {};
    v8f acc1 = {};
    const int mrow  = tr * 16 + (lane & 15);
    const int nrow0 = (tcg * 2 + 0) * 16 + (lane & 15);
    const int nrow1 = (tcg * 2 + 1) * 16 + (lane & 15);
    const int khb   = (lane < 16) ? 0 : 8;   // CDNA5 16-bit A/B lane K-offset

    const int KT = K >> 5;
    for (int kt = 0; kt < KT; ++kt) {
        __syncthreads();
        const int cur = kt & 1;
        const int nxt = cur ^ 1;
        const bool more = (kt + 1) < KT;
        if (more) {
            const int k0 = (kt + 1) << 5;
            ra = *(const v8h*)(A + aoff + k0);
            rb = *(const v8h*)(W + boff + k0);
        }
        // fragments: lanes 0-15 K=0..7,16..23 ; lanes 16-31 K=8..15,24..31
        const _Float16* pa  = &As[cur][mrow  * 32];
        const _Float16* pb0 = &Bs[cur][nrow0 * 32];
        const _Float16* pb1 = &Bs[cur][nrow1 * 32];
        v8h alo = *(const v8h*)(pa  + khb),      ahi = *(const v8h*)(pa  + 16 + khb);
        v8h b0l = *(const v8h*)(pb0 + khb),      b0h = *(const v8h*)(pb0 + 16 + khb);
        v8h b1l = *(const v8h*)(pb1 + khb),      b1h = *(const v8h*)(pb1 + 16 + khb);
        v16h a  = __builtin_shufflevector(alo, ahi, 0,1,2,3,4,5,6,7,8,9,10,11,12,13,14,15);
        v16h b0 = __builtin_shufflevector(b0l, b0h, 0,1,2,3,4,5,6,7,8,9,10,11,12,13,14,15);
        v16h b1 = __builtin_shufflevector(b1l, b1h, 0,1,2,3,4,5,6,7,8,9,10,11,12,13,14,15);
        acc0 = __builtin_amdgcn_wmma_f32_16x16x32_f16(false, a, false, b0, (short)0, acc0, false, false);
        acc1 = __builtin_amdgcn_wmma_f32_16x16x32_f16(false, a, false, b1, (short)0, acc1, false, false);
        if (more) {
            *(v8h*)(&As[nxt][sidx]) = ra;
            *(v8h*)(&Bs[nxt][sidx]) = rb;
        }
    }

    // D layout: VGPR r -> M = r + 8*(lane>=16), N = lane&15
    const int mbase = m0 + tr * 16 + ((lane >= 16) ? 8 : 0);
    const int nc0   = n0 + (tcg * 2 + 0) * 16 + (lane & 15);
    const int nc1   = n0 + (tcg * 2 + 1) * 16 + (lane & 15);
    const float bv0 = (bias && nc0 < N) ? bias[nc0] : 0.0f;
    const float bv1 = (bias && nc1 < N) ? bias[nc1] : 0.0f;
#pragma unroll
    for (int r = 0; r < 8; ++r) {
        int m = mbase + r;
        if (m < M) {
            if (nc0 < N) C[(size_t)m * N + nc0] = acc0[r] + bv0;
            if (nc1 < N) C[(size_t)m * N + nc1] = acc1[r] + bv1;
        }
    }
}

// ========================================================================
// input projection: h[b,t,d] = sum_f x[b,f,t] * W[d,f] + bias[d]
// ========================================================================
__global__ __launch_bounds__(256)
void k_input_proj(const float* __restrict__ x, const float* __restrict__ W,
                  const float* __restrict__ bias, float* __restrict__ h)
{
    int idx = blockIdx.x * blockDim.x + threadIdx.x;
    if (idx >= NROW * DMODEL) return;
    int d  = idx & (DMODEL - 1);
    int bt = idx >> 8;
    int t  = bt & (TLEN - 1);
    int bb = bt >> 10;
    const float* xp = x + (size_t)bb * INFEAT * TLEN + t;
    const float* wp = W + d * INFEAT;
    float acc = bias[d];
#pragma unroll
    for (int f = 0; f < INFEAT; ++f) acc += xp[(size_t)f * TLEN] * wp[f];
    h[idx] = acc;
}

// in-place reverse along T for a (B, T, D) buffer
__global__ __launch_bounds__(256)
void k_flip(float* __restrict__ buf, int D)
{
    int idx = blockIdx.x * blockDim.x + threadIdx.x;
    int total = BATCH * (TLEN / 2) * D;
    if (idx >= total) return;
    int d  = idx % D;
    int r  = idx / D;
    int t  = r % (TLEN / 2);
    int bb = r / (TLEN / 2);
    size_t i0 = ((size_t)bb * TLEN + t) * D + d;
    size_t i1 = ((size_t)bb * TLEN + (TLEN - 1 - t)) * D + d;
    float a = buf[i0], b = buf[i1];
    buf[i0] = b; buf[i1] = a;
}

// out = a + b   (b == nullptr -> copy)
__global__ __launch_bounds__(256)
void k_add(float* __restrict__ out, const float* __restrict__ a,
           const float* __restrict__ b, int n)
{
    int idx = blockIdx.x * blockDim.x + threadIdx.x;
    if (idx >= n) return;
    out[idx] = b ? (a[idx] + b[idx]) : a[idx];
}

// wave-per-row LayerNorm (D multiple of 32); writes f32 and/or f16 outputs
__global__ __launch_bounds__(256)
void k_layernorm(const float* __restrict__ in, const float* __restrict__ w,
                 const float* __restrict__ b, float* __restrict__ out,
                 _Float16* __restrict__ out16, int D, int rows)
{
    int wid  = (blockIdx.x * blockDim.x + threadIdx.x) >> 5;
    int lane = threadIdx.x & 31;
    if (wid >= rows) return;
    const float* rp = in + (size_t)wid * D;
    int per = D >> 5;
    float s = 0.f, s2 = 0.f;
    for (int j = 0; j < per; ++j) {
        float v = rp[lane + j * 32];
        s += v; s2 += v * v;
    }
    for (int m = 16; m >= 1; m >>= 1) {
        s  += __shfl_xor(s,  m, 32);
        s2 += __shfl_xor(s2, m, 32);
    }
    float mu  = s / (float)D;
    float var = s2 / (float)D - mu * mu;
    float inv = rsqrtf(var + EPSV);
    for (int j = 0; j < per; ++j) {
        int c = lane + j * 32;
        float v = (rp[c] - mu) * inv * w[c] + b[c];
        if (out)   out[(size_t)wid * D + c]   = v;
        if (out16) out16[(size_t)wid * D + c] = (_Float16)v;
    }
}

// dt = softplus(zx[..., 1056+h] + dt_bias[h])
__global__ __launch_bounds__(256)
void k_dt(const float* __restrict__ zx, const float* __restrict__ dtb,
          float* __restrict__ dt)
{
    int idx = blockIdx.x * blockDim.x + threadIdx.x;
    if (idx >= NROW * NHEADS) return;
    int h   = idx & (NHEADS - 1);
    int row = idx >> 3;
    float v = zx[(size_t)row * DINPROJ + (DINNER + CONVDIM) + h] + dtb[h];
    dt[idx] = (v > 20.f) ? v : log1pf(__expf(v));
}

// depthwise causal conv (K=4) + SiLU over the xBC slice of zx
__global__ __launch_bounds__(256)
void k_conv(const float* __restrict__ zx, const float* __restrict__ cw,
            const float* __restrict__ cb, float* __restrict__ out)
{
    int idx = blockIdx.x * blockDim.x + threadIdx.x;
    if (idx >= NROW * CONVDIM) return;
    int c   = idx % CONVDIM;
    int row = idx / CONVDIM;
    int t   = row & (TLEN - 1);
    int bb  = row >> 10;
    float acc = cb[c];
#pragma unroll
    for (int k = 0; k < DCONV; ++k) {
        int tt = t + k - (DCONV - 1);
        if (tt >= 0)
            acc += zx[((size_t)(bb * TLEN + tt)) * DINPROJ + DINNER + c] * cw[c * DCONV + k];
    }
    out[idx] = siluf(acc);
}

// ========================================================================
// Sequential SSM scan.  1 block per (batch, head); 256 threads.
// thread: n = tid&15, pgroup = tid>>4; holds 4 state elems (p = pg + 16j).
// y[p] reduced over the 16 n-lanes with shfl_xor(width 16).
// ========================================================================
__global__ __launch_bounds__(256)
void k_scan(const float* __restrict__ xBCc, const float* __restrict__ dt,
            const float* __restrict__ A_log, const float* __restrict__ Dv,
            float* __restrict__ y)
{
    int bh = blockIdx.x;
    int bb = bh >> 3;
    int h  = bh & 7;
    int tid = threadIdx.x;
    int n  = tid & 15;
    int pg = tid >> 4;
    float A  = -__expf(A_log[h]);
    float Dh = Dv[h];
    float hs[4] = {0.f, 0.f, 0.f, 0.f};

    for (int t = 0; t < TLEN; ++t) {
        size_t row = (size_t)(bb * TLEN + t);
        const float* xr = xBCc + row * CONVDIM;
        float dtv = dt[row * NHEADS + h];
        float dA  = __expf(dtv * A);
        float Bt  = xr[DINNER + n];
        float Ct  = xr[DINNER + DSTATE + n];
        float dtB = dtv * Bt;
        float xv[4];
#pragma unroll
        for (int j = 0; j < 4; ++j) xv[j] = xr[h * HEADDIM + pg + 16 * j];
#pragma unroll
        for (int j = 0; j < 4; ++j) {
            hs[j] = hs[j] * dA + dtB * xv[j];
            float yv = hs[j] * Ct;
            for (int m = 8; m >= 1; m >>= 1) yv += __shfl_xor(yv, m, 16);
            if (n == 0) y[row * DINNER + h * HEADDIM + pg + 16 * j] = yv + Dh * xv[j];
        }
    }
}

// y16 = f16( RMSNorm(y * silu(z)) * w )   (wave per row, D = 512)
__global__ __launch_bounds__(256)
void k_gatenorm(const float* __restrict__ y, const float* __restrict__ zx,
                const float* __restrict__ w, _Float16* __restrict__ y16)
{
    int wid  = (blockIdx.x * blockDim.x + threadIdx.x) >> 5;
    int lane = threadIdx.x & 31;
    if (wid >= NROW) return;
    const float* yr = y  + (size_t)wid * DINNER;
    const float* zr = zx + (size_t)wid * DINPROJ;   // z = [0:512]
    float vals[16];
    float s2 = 0.f;
#pragma unroll
    for (int j = 0; j < 16; ++j) {
        int c = lane + j * 32;
        float v = yr[c] * siluf(zr[c]);
        vals[j] = v; s2 += v * v;
    }
    for (int m = 16; m >= 1; m >>= 1) s2 += __shfl_xor(s2, m, 32);
    float inv = rsqrtf(s2 / (float)DINNER + EPSV);
#pragma unroll
    for (int j = 0; j < 16; ++j) {
        int c = lane + j * 32;
        y16[(size_t)wid * DINNER + c] = (_Float16)(vals[j] * inv * w[c]);
    }
}

// g16 = f16( y1 * silu(gate) ) from yy (width 512 -> 256)
__global__ __launch_bounds__(256)
void k_swiglu(const float* __restrict__ yy, _Float16* __restrict__ g16)
{
    int idx = blockIdx.x * blockDim.x + threadIdx.x;
    if (idx >= NROW * DINT) return;
    int c   = idx & (DINT - 1);
    int row = idx >> 8;
    float a  = yy[(size_t)row * (2 * DINT) + c];
    float gt = yy[(size_t)row * (2 * DINT) + DINT + c];
    g16[idx] = (_Float16)(a * siluf(gt));
}

// mean over T: (B, T, 256) -> (B, 256)
__global__ __launch_bounds__(256)
void k_mean(const float* __restrict__ in, float* __restrict__ out)
{
    int bb = blockIdx.x;
    int d  = threadIdx.x;
    float s = 0.f;
    for (int t = 0; t < TLEN; ++t)
        s += in[((size_t)bb * TLEN + t) * DMODEL + d];
    out[bb * DMODEL + d] = s * (1.0f / (float)TLEN);
}

// classifier head: (B,256) @ (10,256)^T + b -> (B,10)
__global__ __launch_bounds__(128)
void k_head(const float* __restrict__ mean, const float* __restrict__ fw,
            const float* __restrict__ fb, float* __restrict__ out)
{
    int idx = threadIdx.x;
    if (idx >= BATCH * NCLS) return;
    int bb = idx / NCLS, c = idx % NCLS;
    float s = fb[c];
    for (int k = 0; k < DMODEL; ++k) s += mean[bb * DMODEL + k] * fw[c * DMODEL + k];
    out[idx] = s;
}

// ---------------- workspace layout ----------------
// fp32 region (floats):
#define OFF_H     ((size_t)0)
#define OFF_RESID (OFF_H     + (size_t)NROW * DMODEL)
#define OFF_HN    (OFF_RESID + (size_t)NROW * DMODEL)   // f32 LN out (final LN only)
#define OFF_HM    (OFF_HN    + (size_t)NROW * DMODEL)
#define OFF_Y     (OFF_HM    + (size_t)NROW * DMODEL)   // scan out / fc1 out (8192x512)
#define OFF_ZX    (OFF_Y     + (size_t)NROW * DINNER)
#define OFF_XBC   (OFF_ZX    + (size_t)NROW * DINPROJ)
#define OFF_DT    (OFF_XBC   + (size_t)NROW * CONVDIM)
#define OFF_MEAN  (OFF_DT    + (size_t)NROW * NHEADS)
#define OFF_F16   (OFF_MEAN  + (size_t)2048)            // start of f16 region (floats)
// f16 region (halves, relative to f16 base):
#define H16_HN    ((size_t)0)                               // 8192*256
#define H16_Y     (H16_HN  + (size_t)NROW * DMODEL)         // 8192*512
#define H16_G     (H16_Y   + (size_t)NROW * DINNER)         // 8192*256
#define H16_WIN   (H16_G   + (size_t)NROW * DMODEL)         // 4*1064*256
#define H16_WOP   (H16_WIN + (size_t)LAYERS * DINPROJ * DMODEL)
#define H16_WF1   (H16_WOP + (size_t)LAYERS * DMODEL * DINNER)
#define H16_WF2   (H16_WF1 + (size_t)LAYERS * 2 * DINT * DMODEL)
// total: ~103 MB fp32 + ~22 MB f16 of d_ws

extern "C" void kernel_launch(void* const* d_in, const int* in_sizes, int n_in,
                              void* d_out, int out_size, void* d_ws, size_t ws_size,
                              hipStream_t stream)
{
    const float* x    = (const float*)d_in[0];
    const float* ipw  = (const float*)d_in[1];
    const float* ipb  = (const float*)d_in[2];
    const float* n1w  = (const float*)d_in[3];
    const float* n1b  = (const float*)d_in[4];
    const float* inw  = (const float*)d_in[5];
    const float* cw   = (const float*)d_in[6];
    const float* cb   = (const float*)d_in[7];
    const float* dtb  = (const float*)d_in[8];
    const float* alog = (const float*)d_in[9];
    const float* Dp   = (const float*)d_in[10];
    const float* snw  = (const float*)d_in[11];
    const float* opw  = (const float*)d_in[12];
    const float* n2w  = (const float*)d_in[13];
    const float* n2b  = (const float*)d_in[14];
    const float* f1w  = (const float*)d_in[15];
    const float* f1b  = (const float*)d_in[16];
    const float* f2w  = (const float*)d_in[17];
    const float* f2b  = (const float*)d_in[18];
    const float* nfw  = (const float*)d_in[19];
    const float* nfb  = (const float*)d_in[20];
    const float* fcw  = (const float*)d_in[21];
    const float* fcb  = (const float*)d_in[22];

    float* ws    = (float*)d_ws;
    float* h     = ws + OFF_H;
    float* resid = ws + OFF_RESID;
    float* hn    = ws + OFF_HN;
    float* hm    = ws + OFF_HM;
    float* y     = ws + OFF_Y;
    float* zx    = ws + OFF_ZX;
    float* xbc   = ws + OFF_XBC;
    float* dtbuf = ws + OFF_DT;
    float* meanb = ws + OFF_MEAN;

    _Float16* f16base = (_Float16*)(ws + OFF_F16);
    _Float16* hn16  = f16base + H16_HN;
    _Float16* y16   = f16base + H16_Y;
    _Float16* g16   = f16base + H16_G;
    _Float16* win16 = f16base + H16_WIN;
    _Float16* wop16 = f16base + H16_WOP;
    _Float16* wf116 = f16base + H16_WF1;
    _Float16* wf216 = f16base + H16_WF2;

    const int nD  = NROW * DMODEL;
    const int TPB = 256;
    const int gD  = (nD + TPB - 1) / TPB;
    const int gFlipD = (BATCH * (TLEN / 2) * DMODEL + TPB - 1) / TPB;
    const dim3 gInProj((DINPROJ + 63) / 64, NROW / 64);   // N=1064
    const dim3 gOutProj(DMODEL / 64,        NROW / 64);   // N=256
    const dim3 gFc1((2 * DINT) / 64,        NROW / 64);   // N=512
    const dim3 gFc2(DMODEL / 64,            NROW / 64);   // N=256

    // ---- one-time (per launch) weight conversion to f16 ----
    {
        int nIn = LAYERS * DINPROJ * DMODEL;
        int nOp = LAYERS * DMODEL * DINNER;
        int nF1 = LAYERS * 2 * DINT * DMODEL;
        int nF2 = LAYERS * DMODEL * DINT;
        k_cvt16<<<(nIn / 4 + TPB - 1) / TPB, TPB, 0, stream>>>(inw, win16, nIn);
        k_cvt16<<<(nOp / 4 + TPB - 1) / TPB, TPB, 0, stream>>>(opw, wop16, nOp);
        k_cvt16<<<(nF1 / 4 + TPB - 1) / TPB, TPB, 0, stream>>>(f1w, wf116, nF1);
        k_cvt16<<<(nF2 / 4 + TPB - 1) / TPB, TPB, 0, stream>>>(f2w, wf216, nF2);
    }

    // ---- input projection ----
    k_input_proj<<<gD, TPB, 0, stream>>>(x, ipw, ipb, h);

    for (int i = 0; i < LAYERS; ++i) {
        const bool flip = (i & 1);
        if (flip) {
            k_flip<<<gFlipD, TPB, 0, stream>>>(h, DMODEL);
            k_flip<<<gFlipD, TPB, 0, stream>>>(resid, DMODEL);
        }
        if (i == 0) k_add<<<gD, TPB, 0, stream>>>(resid, h, nullptr, nD);
        else        k_add<<<gD, TPB, 0, stream>>>(resid, h, resid,  nD);

        k_layernorm<<<NROW / 8, TPB, 0, stream>>>(resid, n1w + i * DMODEL, n1b + i * DMODEL,
                                                  nullptr, hn16, DMODEL, NROW);
        // in_proj: (8192x256) x (1064x256)^T
        k_gemm_wmma<<<gInProj, TPB, 0, stream>>>(hn16, win16 + (size_t)i * DINPROJ * DMODEL,
                                                 nullptr, zx, NROW, DINPROJ, DMODEL);
        k_dt<<<(NROW * NHEADS + TPB - 1) / TPB, TPB, 0, stream>>>(zx, dtb + i * NHEADS, dtbuf);
        k_conv<<<(NROW * CONVDIM + TPB - 1) / TPB, TPB, 0, stream>>>(
            zx, cw + (size_t)i * CONVDIM * DCONV, cb + i * CONVDIM, xbc);
        k_scan<<<BATCH * NHEADS, TPB, 0, stream>>>(xbc, dtbuf, alog + i * NHEADS,
                                                   Dp + i * NHEADS, y);
        k_gatenorm<<<NROW / 8, TPB, 0, stream>>>(y, zx, snw + i * DINNER, y16);
        // out_proj: (8192x512) x (256x512)^T
        k_gemm_wmma<<<gOutProj, TPB, 0, stream>>>(y16, wop16 + (size_t)i * DMODEL * DINNER,
                                                  nullptr, hm, NROW, DMODEL, DINNER);
        k_add<<<gD, TPB, 0, stream>>>(resid, hm, resid, nD);
        k_layernorm<<<NROW / 8, TPB, 0, stream>>>(resid, n2w + i * DMODEL, n2b + i * DMODEL,
                                                  nullptr, hn16, DMODEL, NROW);
        // fc1: (8192x256) x (512x256)^T + b
        k_gemm_wmma<<<gFc1, TPB, 0, stream>>>(hn16, wf116 + (size_t)i * 2 * DINT * DMODEL,
                                              f1b + i * 2 * DINT, y, NROW, 2 * DINT, DMODEL);
        k_swiglu<<<(NROW * DINT + TPB - 1) / TPB, TPB, 0, stream>>>(y, g16);
        // fc2: (8192x256) x (256x256)^T + b
        k_gemm_wmma<<<gFc2, TPB, 0, stream>>>(g16, wf216 + (size_t)i * DMODEL * DINT,
                                              f2b + i * DMODEL, h, NROW, DMODEL, DINT);
        if (flip) {
            k_flip<<<gFlipD, TPB, 0, stream>>>(h, DMODEL);
            k_flip<<<gFlipD, TPB, 0, stream>>>(resid, DMODEL);
        }
    }

    // ---- final: residual add, LN (f32), mean over T, classifier head ----
    k_add<<<gD, TPB, 0, stream>>>(resid, h, resid, nD);
    k_layernorm<<<NROW / 8, TPB, 0, stream>>>(resid, nfw, nfb, hn, nullptr, DMODEL, NROW);
    k_mean<<<BATCH, DMODEL, 0, stream>>>(hn, meanb);
    k_head<<<1, 128, 0, stream>>>(meanb, fcw, fcb, (float*)d_out);
    (void)in_sizes; (void)n_in; (void)out_size; (void)ws_size;
}
